// MultiSplitCrossAttentionBlockSIIS_39702677684938
// MI455X (gfx1250) — matrix operand order repaired and checked by reading, not compiled, not run on hardware
//
#include <hip/hip_runtime.h>

// ---------------------------------------------------------------------------
// MI455X (gfx1250): bf16 WMMA GEMMs with double-buffered async-to-LDS staging.
// ---------------------------------------------------------------------------

typedef unsigned short u16;
typedef unsigned int   u32;
typedef unsigned long long u64;
typedef __attribute__((ext_vector_type(16))) __bf16 v16bf;
typedef __attribute__((ext_vector_type(8)))  float  v8f;

#define C_DIM   384
#define NQ      8192        // positions per quadrant per batch
#define HWD     32768
#define BATCH   4
#define NHEADS  8
#define HDIM    48

// ---- workspace layout (bytes) ----
#define WS_W     ((size_t)0)                       // 33 * 384*384 bf16
#define WS_XLN   ((size_t)9732096)                 // 4 quads * 4 b * 8192 * 384 bf16
#define WS_QKV   ((size_t)110395392)               // 3 * 4 b * 8192 * 384 bf16 (+256 slack)
#define WS_RNO   ((size_t)185893120)               // 2 * 4 * 384 f32
#define WS_AHAT  ((size_t)185905408)               // 4 b * 8 h * 64*64 bf16
#define WS_O1    ((size_t)186167552)               // 4 b * 8192 * 384 bf16
#define WS_Z     ((size_t)211333376)               // 4 b * 32768 * 384 f32
#define WS_ZBF   ((size_t)412659968)               // same, bf16
#define WS_TOTAL ((size_t)513323264)

__device__ __forceinline__ u16 f2bf(float f) {
  u32 u = __float_as_uint(f);
  u = u + 0x7fffu + ((u >> 16) & 1u);
  return (u16)(u >> 16);
}
__device__ __forceinline__ float bf2f(u16 h) {
  return __uint_as_float(((u32)h) << 16);
}

union FragBF { uint4 q[2]; v16bf v; };

// A-matrix 16x32 bf16 fragment from row-major [16][stride] tile.
// lane L: M = L%16, K chunks [c0..c0+7] and [c0+16..c0+23], c0=(L/16)*8.
__device__ __forceinline__ v16bf frag_a(const u16* base, int stride, int lane) {
  const u16* p = base + (lane & 15) * stride + ((lane >> 4) << 3);
  FragBF f;
  f.q[0] = *(const uint4*)(p);
  f.q[1] = *(const uint4*)(p + 16);
  return f.v;
}
// B-matrix 32x16 bf16 fragment; storage holds B^T row-major: stored[n][k].
// lane L: column n = L%16, K chunks [c0..c0+7] and [c0+16..c0+23], c0=(L/16)*8.
__device__ __forceinline__ v16bf frag_b(const u16* base, int stride, int lane) {
  const u16* p = base + (lane & 15) * stride + ((lane >> 4) << 3);
  FragBF f;
  f.q[0] = *(const uint4*)(p);
  f.q[1] = *(const uint4*)(p + 16);
  return f.v;
}
__device__ __forceinline__ v8f wmma_bf(v16bf a, v16bf b, v8f c) {
  return __builtin_amdgcn_wmma_f32_16x16x32_bf16(false, a, false, b, (short)0, c,
                                                 false, false);
}
__device__ __forceinline__ v8f zero8() {
  v8f z;
#pragma unroll
  for (int i = 0; i < 8; ++i) z[i] = 0.f;
  return z;
}

// ---- CDNA5 async copy global -> LDS (ASYNCcnt), per-lane B128 ----
__device__ __forceinline__ u32 lds_off(const void* p) {
  return (u32)(size_t)p;  // low 32 bits of flat LDS address = LDS byte offset
}
__device__ __forceinline__ void async_ld_b128(u32 ldst, const void* gsrc) {
  asm volatile("global_load_async_to_lds_b128 %0, %1, off"
               :: "v"(ldst), "v"((u64)(size_t)gsrc)
               : "memory");
}
__device__ __forceinline__ void wait_async4() {
  asm volatile("s_wait_asynccnt 0x4" ::: "memory");
}
__device__ __forceinline__ void wait_async0() {
  asm volatile("s_wait_asynccnt 0x0" ::: "memory");
}

// issue one 128x32 bf16 A-tile and B-tile (4 async B128 per thread)
__device__ __forceinline__ void stage_async(const u16* gA, const u16* gB,
                                            u16* lA, u16* lB, int tid,
                                            int m0, int n0, int k0) {
#pragma unroll
  for (int r = 0; r < 2; ++r) {
    int idx = tid * 2 + r;
    int row = idx >> 2, ch = (idx & 3) << 3;
    async_ld_b128(lds_off(lA + row * 32 + ch),
                  gA + (size_t)(m0 + row) * 384 + k0 + ch);
    async_ld_b128(lds_off(lB + row * 32 + ch),
                  gB + (size_t)(n0 + row) * 384 + k0 + ch);
  }
}

// ---------------------------------------------------------------------------
// 0) convert all weight matrices to bf16: [blk][Wq,Wk,Wv,Wo] then conv_w
// ---------------------------------------------------------------------------
__global__ void k_wcvt(const float* wq, const float* wk, const float* wv,
                       const float* wo, const float* cw, u16* wbf) {
  int i = blockIdx.x * 256 + threadIdx.x;
  const int NQKVO = 8 * 4 * 147456;
  if (i < NQKVO) {
    int blk = i / (4 * 147456);
    int r = i - blk * (4 * 147456);
    int which = r / 147456;
    int off = r - which * 147456;
    const float* src = (which == 0) ? wq : (which == 1) ? wk : (which == 2) ? wv : wo;
    wbf[i] = f2bf(src[blk * 147456 + off]);
  } else if (i < NQKVO + 147456) {
    wbf[i] = f2bf(cw[i - NQKVO]);
  }
}

// ---------------------------------------------------------------------------
// 1) LayerNorm per quadrant -> XLN bf16 [quad][b][n][c]; also Z init = x^T
// ---------------------------------------------------------------------------
#define LN_STRIDE 132
__global__ __launch_bounds__(256) void k_ln(const float* x, const float* ln1g,
                                            const float* ln1b, const float* ln2g,
                                            const float* ln2b, u16* xln, float* z) {
  extern __shared__ float sm[];
  float* xs  = sm;                        // [384][LN_STRIDE]
  float* mu  = sm + 384 * LN_STRIDE;      // [128]
  float* rsd = mu + 128;                  // [128]
  int tid = threadIdx.x;
  int b = blockIdx.x >> 8;
  int m0 = (blockIdx.x & 255) << 7;

  for (int i = tid; i < 384 * 32; i += 256) {
    int c = i >> 5, col4 = (i & 31) << 2;
    float4 v = *(const float4*)(x + ((size_t)b * 384 + c) * HWD + m0 + col4);
    *(float4*)(xs + c * LN_STRIDE + col4) = v;
  }
  __syncthreads();
  if (tid < 128) {
    float s = 0.f, s2 = 0.f;
#pragma unroll 4
    for (int c = 0; c < 384; ++c) {
      float v = xs[c * LN_STRIDE + tid];
      s += v; s2 += v * v;
    }
    float m = s * (1.f / 384.f);
    float var = s2 * (1.f / 384.f) - m * m;
    mu[tid] = m;
    rsd[tid] = rsqrtf(var + 1e-5f);
  }
  __syncthreads();

  int wv = tid >> 5, lane = tid & 31;
  for (int pp = 0; pp < 16; ++pp) {
    int p = wv * 16 + pp;
    int m = m0 + p;
    int hh = m >> 10, ww = (m >> 5) & 31, dd = m & 31;
    int quad = ((hh >> 4) & 1) | (((dd >> 4) & 1) << 1);
    int n = (hh & 15) * 512 + ww * 16 + (dd & 15);
    const float* g = (quad < 2) ? ln1g : ln2g;
    const float* bb = (quad < 2) ? ln1b : ln2b;
    float mm = mu[p], rr = rsd[p];
    float* zr = z + ((size_t)(b * HWD + m)) * 384;
    u16* xr = xln + ((size_t)((quad * 4 + b) * NQ + n)) * 384;
    for (int c = lane; c < 384; c += 32) {
      float v = xs[c * LN_STRIDE + p];
      zr[c] = v;
      xr[c] = f2bf((v - mm) * rr * g[c] + bb[c]);
    }
  }
}

// ---------------------------------------------------------------------------
// shared GEMM body: C[128x128] = A[128xK] * B[128xK]^T, K=384, bf16 WMMA,
// double-buffered async-to-LDS staging.  Accumulators returned to caller.
// ---------------------------------------------------------------------------
#define GEMM_BODY(Xp, Wp, lA, lB)                                              \
  v8f acc[4][2];                                                               \
  _Pragma("unroll") for (int i = 0; i < 4; ++i)                                \
      _Pragma("unroll") for (int j = 0; j < 2; ++j) acc[i][j] = zero8();       \
  stage_async(Xp, Wp, lA[0], lB[0], tid, m0, n0, 0);                           \
  for (int kt = 0; kt < 12; ++kt) {                                            \
    int cur = kt & 1;                                                          \
    if (kt + 1 < 12) {                                                         \
      stage_async(Xp, Wp, lA[cur ^ 1], lB[cur ^ 1], tid, m0, n0,               \
                  (kt + 1) * 32);                                              \
      wait_async4();                                                           \
    } else {                                                                   \
      wait_async0();                                                           \
    }                                                                          \
    __syncthreads();                                                           \
    v16bf af[4], bfv[2];                                                       \
    _Pragma("unroll") for (int i = 0; i < 4; ++i)                              \
        af[i] = frag_a(lA[cur] + (wm * 64 + i * 16) * 32, 32, lane);           \
    _Pragma("unroll") for (int j = 0; j < 2; ++j)                              \
        bfv[j] = frag_b(lB[cur] + (wn * 32 + j * 16) * 32, 32, lane);          \
    _Pragma("unroll") for (int i = 0; i < 4; ++i)                              \
        _Pragma("unroll") for (int j = 0; j < 2; ++j)                          \
            acc[i][j] = wmma_bf(af[i], bfv[j], acc[i][j]);                     \
    __syncthreads();                                                           \
  }

// ---------------------------------------------------------------------------
// 2) QKV projection GEMM: Out[pr][b] = XLN[quad][b] @ W^T
// ---------------------------------------------------------------------------
__global__ __launch_bounds__(256) void k_proj(const u16* xln, const u16* wbf,
                                              u16* qkv, int blk, int q1, int q2) {
  __shared__ u16 As[2][128 * 32];
  __shared__ u16 Bs[2][128 * 32];
  int m0 = blockIdx.x * 128, n0 = blockIdx.y * 128;
  int pr = blockIdx.z >> 2, b = blockIdx.z & 3;
  int quad = (pr == 0) ? q1 : q2;
  const u16* X = xln + (size_t)(quad * 4 + b) * NQ * 384;
  const u16* Wm = wbf + (size_t)(blk * 4 + pr) * 147456;
  u16* Out = qkv + (size_t)(pr * 4 + b) * NQ * 384;
  int tid = threadIdx.x, lane = tid & 31, wv = tid >> 5;
  int wm = wv & 1, wn = wv >> 1;

  GEMM_BODY(X, Wm, As, Bs)

#pragma unroll
  for (int i = 0; i < 4; ++i)
#pragma unroll
    for (int j = 0; j < 2; ++j)
#pragma unroll
      for (int r = 0; r < 8; ++r) {
        int mm = m0 + wm * 64 + i * 16 + r + ((lane >> 4) << 3);
        int nn = n0 + wn * 32 + j * 16 + (lane & 15);
        Out[(size_t)mm * 384 + nn] = f2bf(acc[i][j][r]);
      }
}

// ---------------------------------------------------------------------------
// 3) column sum-of-squares over N for Q and K -> rnorm = 1/max(||col||,1e-12)
// ---------------------------------------------------------------------------
__global__ __launch_bounds__(256) void k_ssq(const u16* qkv, float* rno) {
  __shared__ float red[256];
  int tid = threadIdx.x;
  int cg = blockIdx.x, b = blockIdx.y, qk = blockIdx.z;
  int half = tid >> 7, cl = tid & 127;
  int c = cg * 128 + cl;
  const u16* base = qkv + (size_t)(qk * 4 + b) * NQ * 384;
  float acc = 0.f;
  int nbeg = half * 4096;
  for (int n = nbeg; n < nbeg + 4096; ++n) {
    float v = bf2f(base[(size_t)n * 384 + c]);
    acc += v * v;
  }
  red[tid] = acc;
  __syncthreads();
  if (tid < 128) {
    float s = red[tid] + red[tid + 128];
    rno[(qk * 4 + b) * 384 + c] = 1.f / fmaxf(sqrtf(s), 1e-12f);
  }
}

// ---------------------------------------------------------------------------
// 4) S = Q_h^T K_h (48x48, K=8192) via WMMA; scale; softmax; Ahat [64][64]
// ---------------------------------------------------------------------------
__global__ __launch_bounds__(256) void k_attn(const u16* qkv, const float* rno,
                                              const float* temp, u16* ahat, int blk) {
  extern __shared__ char smc[];
  u16* Qs = (u16*)smc;                    // [48][512] transposed
  u16* Ks = Qs + 48 * 512;                // [48][512] transposed
  float* Sb = (float*)(Ks + 48 * 512);    // [48][48]
  int h = blockIdx.x, b = blockIdx.y;
  int tid = threadIdx.x, lane = tid & 31, wv = tid >> 5;
  const u16* Q = qkv + (size_t)(0 * 4 + b) * NQ * 384 + h * HDIM;
  const u16* K = qkv + (size_t)(1 * 4 + b) * NQ * 384 + h * HDIM;

  int mt0 = wv / 3, nt0 = wv - mt0 * 3;
  v8f accA = zero8(), accB = zero8();

  for (int ch = 0; ch < 16; ++ch) {
    int n0 = ch * 512;
    __syncthreads();
    for (int idx = tid; idx < 48 * 512; idx += 256) {
      int n = idx / 48;
      int c = idx - n * 48;
      Qs[c * 512 + n] = Q[(size_t)(n0 + n) * 384 + c];
      Ks[c * 512 + n] = K[(size_t)(n0 + n) * 384 + c];
    }
    __syncthreads();
#pragma unroll 4
    for (int kk = 0; kk < 512; kk += 32) {
      v16bf a0 = frag_a(Qs + (mt0 * 16) * 512 + kk, 512, lane);
      v16bf b0 = frag_b(Ks + (nt0 * 16) * 512 + kk, 512, lane);
      accA = wmma_bf(a0, b0, accA);
      if (wv == 0) {
        v16bf a1 = frag_a(Qs + 32 * 512 + kk, 512, lane);
        v16bf b1 = frag_b(Ks + 32 * 512 + kk, 512, lane);
        accB = wmma_bf(a1, b1, accB);
      }
    }
  }
  __syncthreads();
#pragma unroll
  for (int r = 0; r < 8; ++r) {
    int d = mt0 * 16 + r + ((lane >> 4) << 3);
    int e = nt0 * 16 + (lane & 15);
    Sb[d * 48 + e] = accA[r];
    if (wv == 0) {
      int d2 = 32 + r + ((lane >> 4) << 3);
      int e2 = 32 + (lane & 15);
      Sb[d2 * 48 + e2] = accB[r];
    }
  }
  __syncthreads();
  if (tid < 64) {
    u16* ar = ahat + (size_t)(b * 8 + h) * 4096 + tid * 64;
    if (tid < 48) {
      int d = tid;
      float rq = rno[(0 * 4 + b) * 384 + h * HDIM + d];
      float tt = temp[blk * NHEADS + h];
      float rowv[48];
      float mx = -3.4e38f;
#pragma unroll
      for (int e = 0; e < 48; ++e) {
        float rk = rno[(1 * 4 + b) * 384 + h * HDIM + e];
        float v = Sb[d * 48 + e] * rq * rk * tt;
        rowv[e] = v;
        mx = fmaxf(mx, v);
      }
      float sum = 0.f;
#pragma unroll
      for (int e = 0; e < 48; ++e) {
        rowv[e] = expf(rowv[e] - mx);
        sum += rowv[e];
      }
      float inv = 1.f / sum;
#pragma unroll
      for (int e = 0; e < 48; ++e) ar[e] = f2bf(rowv[e] * inv);
#pragma unroll
      for (int e = 48; e < 64; ++e) ar[e] = 0;
    } else {
#pragma unroll
      for (int e = 0; e < 64; ++e) ar[e] = 0;
    }
  }
}

// ---------------------------------------------------------------------------
// 5) per-head O1 = V_h @ Ahat^T  (M=8192, N=48, K=64 padded)
// ---------------------------------------------------------------------------
__global__ __launch_bounds__(256) void k_outhead(const u16* qkv, const u16* ahat,
                                                 u16* o1) {
  int m0 = blockIdx.x * 128, h = blockIdx.y, b = blockIdx.z;
  int tid = threadIdx.x, lane = tid & 31, wv = tid >> 5;
  const u16* V = qkv + (size_t)(2 * 4 + b) * NQ * 384;
  const u16* Ah = ahat + (size_t)(b * 8 + h) * 4096;
  v16bf bfr[3][2];
#pragma unroll
  for (int nt = 0; nt < 3; ++nt)
#pragma unroll
    for (int ks = 0; ks < 2; ++ks)
      bfr[nt][ks] = frag_b(Ah + (nt * 16) * 64 + ks * 32, 64, lane);
  v8f acc[3];
#pragma unroll
  for (int nt = 0; nt < 3; ++nt) acc[nt] = zero8();
#pragma unroll
  for (int ks = 0; ks < 2; ++ks) {
    v16bf af = frag_a(V + (size_t)(m0 + wv * 16) * 384 + h * HDIM + ks * 32, 384, lane);
#pragma unroll
    for (int nt = 0; nt < 3; ++nt) acc[nt] = wmma_bf(af, bfr[nt][ks], acc[nt]);
  }
#pragma unroll
  for (int nt = 0; nt < 3; ++nt)
#pragma unroll
    for (int r = 0; r < 8; ++r) {
      int mm = m0 + wv * 16 + r + ((lane >> 4) << 3);
      int cc = h * HDIM + nt * 16 + (lane & 15);
      o1[((size_t)(b * NQ) + mm) * 384 + cc] = f2bf(acc[nt][r]);
    }
}

// ---------------------------------------------------------------------------
// 6) Wo GEMM + epilogue: Z[b][mz][c] += gamma[c] * (O1@Wo^T + bo)[n][c]
// ---------------------------------------------------------------------------
__global__ __launch_bounds__(256) void k_wo(const u16* o1, const u16* wbf,
                                            const float* bo, const float* gamma,
                                            float* z, int blk) {
  __shared__ u16 As[2][128 * 32];
  __shared__ u16 Bs[2][128 * 32];
  int m0 = blockIdx.x * 128, n0 = blockIdx.y * 128, b = blockIdx.z;
  const u16* X = o1 + (size_t)b * NQ * 384;
  const u16* Wm = wbf + (size_t)(blk * 4 + 3) * 147456;
  int tid = threadIdx.x, lane = tid & 31, wv = tid >> 5;
  int wm = wv & 1, wn = wv >> 1;
  int zq = blk & 3;

  GEMM_BODY(X, Wm, As, Bs)

#pragma unroll
  for (int i = 0; i < 4; ++i)
#pragma unroll
    for (int j = 0; j < 2; ++j)
#pragma unroll
      for (int r = 0; r < 8; ++r) {
        int ml = m0 + wm * 64 + i * 16 + r + ((lane >> 4) << 3);
        int cg = n0 + wn * 32 + j * 16 + (lane & 15);
        size_t zi = ((size_t)(b * HWD + zq * NQ + ml)) * 384 + cg;
        z[zi] += gamma[cg] * (acc[i][j][r] + bo[cg]);
      }
}

// ---------------------------------------------------------------------------
// 7) Z -> bf16
// ---------------------------------------------------------------------------
__global__ void k_zbf(const float* z, u16* zbf) {
  size_t i = ((size_t)blockIdx.x * 256 + threadIdx.x) * 4;
  if (i < (size_t)BATCH * HWD * 384) {
    float4 v = *(const float4*)(z + i);
    u32 lo = (u32)f2bf(v.x) | ((u32)f2bf(v.y) << 16);
    u32 hi = (u32)f2bf(v.z) | ((u32)f2bf(v.w) << 16);
    uint2 pk; pk.x = lo; pk.y = hi;
    *(uint2*)(zbf + i) = pk;
  }
}

// ---------------------------------------------------------------------------
// 8) final: out[b][o][m] = Z[b][m][o] + (Z @ conv_w^T)[m][o] + conv_b[o]
//    A = conv_w rows (o), B = Zbf rows (m) -> stores m-contiguous.
// ---------------------------------------------------------------------------
__global__ __launch_bounds__(256) void k_conv(const u16* zbf, const float* z,
                                              const u16* wcv, const float* cb,
                                              float* out) {
  __shared__ u16 As[2][128 * 32];
  __shared__ u16 Bs[2][128 * 32];
  int m0_ = blockIdx.x * 128, o0 = blockIdx.y * 128, b = blockIdx.z;
  int tid = threadIdx.x, lane = tid & 31, wv = tid >> 5;
  int wm = wv & 1, wn = wv >> 1;
  const u16* Av = wcv;                               // A rows = o
  const u16* Bv = zbf + (size_t)(b * HWD) * 384;     // B rows = m
  int m0 = o0;    // GEMM_BODY "m" = output rows = o
  int n0 = m0_;   // GEMM_BODY "n" = output cols = m

  GEMM_BODY(Av, Bv, As, Bs)

#pragma unroll
  for (int i = 0; i < 4; ++i)
#pragma unroll
    for (int j = 0; j < 2; ++j)
#pragma unroll
      for (int r = 0; r < 8; ++r) {
        int oo = o0 + wm * 64 + i * 16 + r + ((lane >> 4) << 3);
        int mm = m0_ + wn * 32 + j * 16 + (lane & 15);
        float res = z[((size_t)(b * HWD) + mm) * 384 + oo];
        out[((size_t)(b * 384) + oo) * HWD + mm] = acc[i][j][r] + res + cb[oo];
      }
}

// ---------------------------------------------------------------------------
extern "C" void kernel_launch(void* const* d_in, const int* in_sizes, int n_in,
                              void* d_out, int out_size, void* d_ws, size_t ws_size,
                              hipStream_t stream) {
  (void)in_sizes; (void)n_in; (void)out_size;
  if (ws_size < WS_TOTAL) return;  // workspace too small; cannot run safely

  const float* x     = (const float*)d_in[0];
  const float* ln1g  = (const float*)d_in[1];
  const float* ln1b  = (const float*)d_in[2];
  const float* ln2g  = (const float*)d_in[3];
  const float* ln2b  = (const float*)d_in[4];
  const float* gamma = (const float*)d_in[5];
  const float* Wq    = (const float*)d_in[6];
  const float* Wk    = (const float*)d_in[7];
  const float* Wv    = (const float*)d_in[8];
  const float* Wo    = (const float*)d_in[9];
  const float* bo    = (const float*)d_in[10];
  const float* temp  = (const float*)d_in[11];
  const float* convw = (const float*)d_in[12];
  const float* convb = (const float*)d_in[13];
  float* out = (float*)d_out;

  char* ws = (char*)d_ws;
  u16*   Wbf  = (u16*)(ws + WS_W);
  u16*   XLN  = (u16*)(ws + WS_XLN);
  u16*   QKV  = (u16*)(ws + WS_QKV);
  float* RNO  = (float*)(ws + WS_RNO);
  u16*   AHAT = (u16*)(ws + WS_AHAT);
  u16*   O1   = (u16*)(ws + WS_O1);
  float* Z    = (float*)(ws + WS_Z);
  u16*   ZBF  = (u16*)(ws + WS_ZBF);

  k_wcvt<<<19008, 256, 0, stream>>>(Wq, Wk, Wv, Wo, convw, Wbf);
  size_t ln_smem = (size_t)(384 * LN_STRIDE + 256) * 4;
  k_ln<<<1024, 256, ln_smem, stream>>>(x, ln1g, ln1b, ln2g, ln2b, XLN, Z);

  // quadrant ids: 0=sup_mac 1=inf_mac 2=sup_nerve 3=inf_nerve
  const int q1[8] = {0, 1, 2, 3, 2, 0, 3, 1};  // query sources
  const int q2[8] = {1, 0, 3, 2, 0, 2, 1, 3};  // key/value sources
  size_t attn_smem = 2 * 48 * 512 * sizeof(u16) + 48 * 48 * sizeof(float);

  for (int blk = 0; blk < 8; ++blk) {
    k_proj<<<dim3(64, 3, 12), 256, 0, stream>>>(XLN, Wbf, QKV, blk, q1[blk], q2[blk]);
    k_ssq<<<dim3(3, 4, 2), 256, 0, stream>>>(QKV, RNO);
    k_attn<<<dim3(8, 4), 256, attn_smem, stream>>>(QKV, RNO, temp, AHAT, blk);
    k_outhead<<<dim3(64, 8, 4), 256, 0, stream>>>(QKV, AHAT, O1);
    k_wo<<<dim3(64, 3, 4), 256, 0, stream>>>(O1, Wbf, bo + blk * 384, gamma, Z, blk);
  }

  k_zbf<<<49152, 256, 0, stream>>>(Z, ZBF);
  k_conv<<<dim3(256, 3, 4), 256, 0, stream>>>(ZBF, Z, Wbf + (size_t)32 * 147456,
                                              convb, out);
}
